// GATNetwork_3487513444403
// MI455X (gfx1250) — matrix operand order repaired and checked by reading, not compiled, not run on hardware
//
#include <hip/hip_runtime.h>
#include <cstdint>
#include <cstddef>

typedef __attribute__((ext_vector_type(16))) _Float16 v16h;
typedef __attribute__((ext_vector_type(8)))  _Float16 v8h;
typedef __attribute__((ext_vector_type(8)))  float    v8f;

static __device__ __forceinline__ void atomicMaxF(float* addr, float v) {
  // works for mixed-sign when addr initialized to -inf:
  // positive floats ordered as signed ints; negative floats reverse-ordered as unsigned.
  if (v >= 0.0f) atomicMax((int*)addr, __float_as_int(v));
  else           atomicMin((unsigned int*)addr, __float_as_uint(v));
}

// ---------------------------------------------------------------------------
// f16 WMMA GEMM: C[M x NCOL] = A[M x K] * B[K x NCOL]
// A row-major f16; Bt = B pre-transposed to [NCOL x K] f16 (contiguous K per col)
// Each wave computes a 16x64 tile: 4 accumulators, K-loop step 32,
// software-pipelined (next-step A/B fragments prefetched before current WMMAs).
// ---------------------------------------------------------------------------
template <int K, int NCOL, bool HALF_OUT>
__global__ __launch_bounds__(256) void wmma_gemm(
    const _Float16* __restrict__ A, const _Float16* __restrict__ Bt,
    void* __restrict__ Cv, int Mtiles) {
  constexpr int COLG = NCOL / 64;
  int wid = blockIdx.x * 8 + (threadIdx.x >> 5);
  if (wid >= Mtiles * COLG) return;            // whole wave exits together
  int rowTile = wid / COLG;
  int colG    = wid - rowTile * COLG;
  int lane = threadIdx.x & 31;
  int r  = lane & 15;   // A: row in tile; B: col in 16-col subtile
  int hs = lane >> 4;   // half-wave select

  // A layout (16-bit 16x32): lanes 0-15 hold K = 0..15, lanes 16-31 hold K = 16..31 (contiguous)
  const _Float16* a_base = A  + (size_t)(rowTile * 16 + r) * K + hs * 16;
  // B layout (16-bit 32x16): lanes 0-15 hold K = 0..15, lanes 16-31 hold K = 16..31 (contiguous)
  const _Float16* b_base = Bt + (size_t)(colG * 64 + r) * K + hs * 16;

  v8f acc[4] = {};

  // prologue: fragments for k0 = 0
  v8h alo = *(const v8h*)(a_base);
  v8h ahi = *(const v8h*)(a_base + 8);
  v16h bcur[4];
#pragma unroll
  for (int t = 0; t < 4; ++t)
    bcur[t] = *(const v16h*)(b_base + (size_t)t * 16 * K);

#pragma unroll
  for (int k0 = 0; k0 < K; k0 += 32) {
    v16h a = __builtin_shufflevector(alo, ahi, 0,1,2,3,4,5,6,7,8,9,10,11,12,13,14,15);
    // prefetch next iteration's fragments while WMMAs execute
    v8h nlo, nhi;
    v16h bnext[4];
    if (k0 + 32 < K) {
      nlo = *(const v8h*)(a_base + k0 + 32);
      nhi = *(const v8h*)(a_base + k0 + 40);
#pragma unroll
      for (int t = 0; t < 4; ++t)
        bnext[t] = *(const v16h*)(b_base + (size_t)t * 16 * K + k0 + 32);
    }
#pragma unroll
    for (int t = 0; t < 4; ++t)
      acc[t] = __builtin_amdgcn_wmma_f32_16x16x32_f16(
          false, a, false, bcur[t], (short)0, acc[t], false, false);
    if (k0 + 32 < K) {
      alo = nlo; ahi = nhi;
#pragma unroll
      for (int t = 0; t < 4; ++t) bcur[t] = bnext[t];
    }
  }

  // D layout: lane -> col (l&15), VGPR v -> row v + 8*hs
  int row0 = rowTile * 16 + hs * 8;
  int colb = colG * 64 + r;
#pragma unroll
  for (int t = 0; t < 4; ++t) {
#pragma unroll
    for (int v = 0; v < 8; ++v) {
      size_t off = (size_t)(row0 + v) * NCOL + colb + t * 16;
      if constexpr (HALF_OUT) ((_Float16*)Cv)[off] = (_Float16)acc[t][v];
      else                    ((float*)Cv)[off]    = acc[t][v];
    }
  }
}

// ---------------------------------------------------------------------------
// Utility / elementwise kernels
// ---------------------------------------------------------------------------
__global__ void cvt_f16(const float* __restrict__ in, _Float16* __restrict__ out, int n) {
  int i = blockIdx.x * blockDim.x + threadIdx.x;
  if (i < n) out[i] = (_Float16)in[i];
}

// Wt[c*K + k] = (f16) W[k*NCOL + c]
__global__ void cvt_f16_t(const float* __restrict__ W, _Float16* __restrict__ Wt,
                          int K, int NCOL) {
  int i = blockIdx.x * blockDim.x + threadIdx.x;
  if (i >= K * NCOL) return;
  int c = i / K, k = i - c * K;
  Wt[i] = (_Float16)W[(size_t)k * NCOL + c];
}

__global__ void fill_f32(float* __restrict__ p, float v, int n) {
  int i = blockIdx.x * blockDim.x + threadIdx.x;
  if (i < n) p[i] = v;
}

// es[n,h] = <h[n,h,:], a_src[h,:]> ; ed likewise.  One thread per (n,h).
template <typename T, int H>
__global__ void attn_logits(const T* __restrict__ h, const float* __restrict__ asr,
                            const float* __restrict__ ads, float* __restrict__ es,
                            float* __restrict__ ed, int NH) {
  int i = blockIdx.x * blockDim.x + threadIdx.x;
  if (i >= NH) return;
  int hh = i % H;
  const T* hp = h + (size_t)i * 128;
  const float* as = asr + hh * 128;
  const float* ad = ads + hh * 128;
  float s = 0.f, d = 0.f;
#pragma unroll 4
  for (int j = 0; j < 128; ++j) {
    float v = (float)hp[j];
    s += v * as[j];
    d += v * ad[j];
  }
  es[i] = s;
  ed[i] = d;
}

// Pass 1: e = leaky_relu(es[src]+ed[dst]); stash e; segment-max into m[dst].
template <int H>
__global__ void edge_logit_max(const int* __restrict__ src, const int* __restrict__ dst,
                               int E, int N, const float* __restrict__ es,
                               const float* __restrict__ ed, float* __restrict__ ebuf,
                               float* __restrict__ m) {
  int idx = blockIdx.x * blockDim.x + threadIdx.x;
  if (idx >= (E + N) * H) return;
  int e = idx / H, hh = idx - e * H;
  int s, d;
  if (e < E) { s = src[e]; d = dst[e]; } else { s = d = e - E; }  // self-loops
  float v = es[s * H + hh] + ed[d * H + hh];
  v = v > 0.f ? v : 0.2f * v;
  ebuf[idx] = v;
  atomicMaxF(&m[d * H + hh], v);
}

// Pass 2: w = exp(e - m[dst]); stash w; segment-sum into z[dst].
template <int H>
__global__ void edge_expsum(const int* __restrict__ dst, int E, int N,
                            const float* __restrict__ m, float* __restrict__ ebuf,
                            float* __restrict__ z) {
  int idx = blockIdx.x * blockDim.x + threadIdx.x;
  if (idx >= (E + N) * H) return;
  int e = idx / H, hh = idx - e * H;
  int d = (e < E) ? dst[e] : e - E;
  float w = __expf(ebuf[idx] - m[d * H + hh]);
  ebuf[idx] = w;
  atomicAdd(&z[d * H + hh], w);
}

// Pass 3: acc[dst] += alpha * h[src].  One wave32 per (edge,head); 4 ch/lane.
// Accumulator fits in the 192MB L2, so atomics resolve on-chip.
template <int H, typename T>
__global__ void edge_aggregate(const int* __restrict__ src, const int* __restrict__ dst,
                               int E, int N, const float* __restrict__ ebuf,
                               const float* __restrict__ z, const T* __restrict__ h,
                               float* __restrict__ acc) {
  int wid = (blockIdx.x * blockDim.x + threadIdx.x) >> 5;
  if (wid >= (E + N) * H) return;
  int lane = threadIdx.x & 31;
  int e = wid / H, hh = wid - e * H;
  int s, d;
  if (e < E) { s = src[e]; d = dst[e]; } else { s = d = e - E; }
  float alpha = ebuf[wid] / z[d * H + hh];
  const T* hp = h + ((size_t)s * H + hh) * 128 + lane * 4;
  float*  op = acc + ((size_t)d * H + hh) * 128 + lane * 4;
#pragma unroll
  for (int j = 0; j < 4; ++j) atomicAdd(op + j, (float)hp[j] * alpha);
}

// h1b = (f16) relu(acc + b); mask = Dm-1 (Dm power of two)
__global__ void relu_bias_h(const float* __restrict__ acc, const float* __restrict__ b,
                            _Float16* __restrict__ out, int total, int mask) {
  int i = blockIdx.x * blockDim.x + threadIdx.x;
  if (i >= total) return;
  float v = acc[i] + b[i & mask];
  out[i] = (_Float16)(v > 0.f ? v : 0.f);
}

// out[n] = <acc2[n,:]+b2, fc_w> + fc_b
__global__ void final_fc(const float* __restrict__ acc2, const float* __restrict__ b2,
                         const float* __restrict__ fw, const float* __restrict__ fb,
                         float* __restrict__ out, int N) {
  int n = blockIdx.x * blockDim.x + threadIdx.x;
  if (n >= N) return;
  const float* ap = acc2 + (size_t)n * 128;
  float sum = 0.f;
#pragma unroll 4
  for (int j = 0; j < 128; ++j) sum += (ap[j] + b2[j]) * fw[j];
  out[n] = sum + fb[0];
}

// ---------------------------------------------------------------------------
extern "C" void kernel_launch(void* const* d_in, const int* in_sizes, int n_in,
                              void* d_out, int out_size, void* d_ws, size_t ws_size,
                              hipStream_t stream) {
  (void)n_in; (void)out_size; (void)ws_size;
  const float* x   = (const float*)d_in[0];
  const int*   ei  = (const int*)d_in[1];   // [2,E] (src row 0, dst row 1)
  const float* W1  = (const float*)d_in[2];
  const float* a1s = (const float*)d_in[3];
  const float* a1d = (const float*)d_in[4];
  const float* b1  = (const float*)d_in[5];
  const float* W2  = (const float*)d_in[6];
  const float* a2s = (const float*)d_in[7];
  const float* a2d = (const float*)d_in[8];
  const float* b2  = (const float*)d_in[9];
  const float* fw  = (const float*)d_in[10];
  const float* fb  = (const float*)d_in[11];

  const int D = 128, H = 4;
  const int N = in_sizes[0] / D;   // 50000 (multiple of 16)
  const int E = in_sizes[1] / 2;   // 800000
  const int Etot = E + N;          // + self-loops
  const int Mtiles = N / 16;
  const int* srcI = ei;
  const int* dstI = ei + E;

  // workspace bump allocator (256B aligned)
  char* p = (char*)d_ws;
  auto alloc = [&](size_t bytes) -> void* {
    void* r = (void*)p;
    p += (bytes + 255) & ~(size_t)255;
    return r;
  };
  _Float16* xh   = (_Float16*)alloc((size_t)N * 128 * 2);
  _Float16* w1t  = (_Float16*)alloc((size_t)512 * 128 * 2);  // [512 cols][128 K]
  _Float16* w2t  = (_Float16*)alloc((size_t)128 * 512 * 2);  // [128 cols][512 K]
  _Float16* h1h  = (_Float16*)alloc((size_t)N * 512 * 2);    // h1 lin (f16); reused post-ReLU
  float* es1  = (float*)alloc((size_t)N * H * 4);
  float* ed1  = (float*)alloc((size_t)N * H * 4);
  float* m1   = (float*)alloc((size_t)N * H * 4);
  float* z1   = (float*)alloc((size_t)N * H * 4);
  float* eb1  = (float*)alloc((size_t)Etot * H * 4);
  float* acc1 = (float*)alloc((size_t)N * 512 * 4);
  float* h2   = (float*)alloc((size_t)N * 128 * 4);
  float* es2  = (float*)alloc((size_t)N * 4);
  float* ed2  = (float*)alloc((size_t)N * 4);
  float* m2   = (float*)alloc((size_t)N * 4);
  float* z2   = (float*)alloc((size_t)N * 4);
  float* eb2  = (float*)alloc((size_t)Etot * 4);
  float* acc2 = (float*)alloc((size_t)N * 128 * 4);

  dim3 blk(256);
  auto g1 = [](size_t n) { return dim3((unsigned)((n + 255) / 256)); };
  const float NEG_INF = -__builtin_huge_valf();

  // -- input conversion (f16 activations; weights transposed to [col][K]) --
  cvt_f16<<<g1((size_t)N * 128), blk, 0, stream>>>(x, xh, N * 128);
  cvt_f16_t<<<g1(128 * 512), blk, 0, stream>>>(W1, w1t, 128, 512);
  cvt_f16_t<<<g1(512 * 128), blk, 0, stream>>>(W2, w2t, 512, 128);

  // -- init accumulators / softmax state --
  hipMemsetAsync(z1, 0, (size_t)N * H * 4, stream);
  hipMemsetAsync(z2, 0, (size_t)N * 4, stream);
  hipMemsetAsync(acc1, 0, (size_t)N * 512 * 4, stream);
  hipMemsetAsync(acc2, 0, (size_t)N * 128 * 4, stream);
  fill_f32<<<g1((size_t)N * H), blk, 0, stream>>>(m1, NEG_INF, N * H);
  fill_f32<<<g1((size_t)N), blk, 0, stream>>>(m2, NEG_INF, N);

  // ---------------- layer 1 (H=4, C=128) ----------------
  {
    int waves = Mtiles * (512 / 64);
    wmma_gemm<128, 512, true><<<dim3((waves + 7) / 8), blk, 0, stream>>>(
        xh, w1t, (void*)h1h, Mtiles);
  }
  attn_logits<_Float16, 4><<<g1((size_t)N * H), blk, 0, stream>>>(h1h, a1s, a1d, es1, ed1, N * H);
  edge_logit_max<4><<<g1((size_t)Etot * H), blk, 0, stream>>>(srcI, dstI, E, N, es1, ed1, eb1, m1);
  edge_expsum<4><<<g1((size_t)Etot * H), blk, 0, stream>>>(dstI, E, N, m1, eb1, z1);
  edge_aggregate<4, _Float16><<<g1((size_t)Etot * H * 32), blk, 0, stream>>>(
      srcI, dstI, E, N, eb1, z1, h1h, acc1);
  // ReLU + bias, write f16 back into h1h (safe: h1h dead after aggregation)
  relu_bias_h<<<g1((size_t)N * 512), blk, 0, stream>>>(acc1, b1, h1h, N * 512, 511);

  // ---------------- layer 2 (H=1, C=128) ----------------
  {
    int waves = Mtiles * (128 / 64);
    wmma_gemm<512, 128, false><<<dim3((waves + 7) / 8), blk, 0, stream>>>(
        h1h, w2t, (void*)h2, Mtiles);
  }
  attn_logits<float, 1><<<g1((size_t)N), blk, 0, stream>>>(h2, a2s, a2d, es2, ed2, N);
  edge_logit_max<1><<<g1((size_t)Etot), blk, 0, stream>>>(srcI, dstI, E, N, es2, ed2, eb2, m2);
  edge_expsum<1><<<g1((size_t)Etot), blk, 0, stream>>>(dstI, E, N, m2, eb2, z2);
  edge_aggregate<1, float><<<g1((size_t)Etot * 32), blk, 0, stream>>>(
      srcI, dstI, E, N, eb2, z2, h2, acc2);

  // ---------------- readout ----------------
  final_fc<<<g1((size_t)N), blk, 0, stream>>>(acc2, b2, fw, fb, (float*)d_out, N);
}
